// SATA_23261542875327
// MI455X (gfx1250) — matrix operand order repaired and checked by reading, not compiled, not run on hardware
//
#include <hip/hip_runtime.h>
#include <hip/hip_bf16.h>
#include <math.h>

// ---------------- CDNA5 WMMA types ----------------
typedef __attribute__((ext_vector_type(16))) __bf16 v16bf;
typedef __attribute__((ext_vector_type(8)))  float  v8f;

#define NNODE 4096
#define NFEAT 256
#define NCLS  16
#define NEDGE 65536
#define NLAYER 4

union FragU { v16bf v; uint4 u[2]; };

__device__ inline __bf16 f2bf(float f) {
  // round-to-nearest-even fp32 -> bf16 (bit trick; avoids relying on cvt codegen)
  union { float f; unsigned u; } c; c.f = f;
  unsigned r = (c.u + 0x7FFFu + ((c.u >> 16) & 1u)) >> 16;
  unsigned short s = (unsigned short)r;
  __bf16 b;
  __builtin_memcpy(&b, &s, 2);
  return b;
}

// ---------------- fragment loaders (wave32 layouts per CDNA5 ISA 7.12.2) ----------------
// A 16x32 bf16: lane L holds row m=L&15; elems 0..7 = K kBase+8h..+7, elems 8..15 = K kBase+16+8h..+7
__device__ inline v16bf load_a_frag(const __bf16* a, int lda, int mBase, int kBase, int lane) {
  int m = lane & 15, h = lane >> 4;
  const __bf16* p = a + (size_t)(mBase + m) * lda + (kBase + 8 * h);
  FragU f;
  f.u[0] = *(const uint4*)(p);
  f.u[1] = *(const uint4*)(p + 16);
  return f.v;
}
// B 32x16 bf16: lane L holds col n=L&15; elems 0..7 = K kBase+8h..+7, elems 8..15 = K kBase+16+8h..+7
// BT is the transposed operand buffer: BT[n][k] = B[k][n], row stride ldbt -> two 16B row loads at k and k+16.
__device__ inline v16bf load_b_frag(const __bf16* bT, int ldbt, int nBase, int kBase, int lane) {
  int n = lane & 15, h = lane >> 4;
  const __bf16* p = bT + (size_t)(nBase + n) * ldbt + (kBase + 8 * h);
  FragU f;
  f.u[0] = *(const uint4*)(p);
  f.u[1] = *(const uint4*)(p + 16);
  return f.v;
}

// C/D f32 16x16: lane L col n=L&15; VGPR v -> row 8*(L>>4)+v
__device__ inline void store_tile(float* C, int ldc, int mBase, int nBase, v8f acc,
                                  int lane, int mode, const float* aux) {
  int n = lane & 15, h = lane >> 4;
#pragma unroll
  for (int v = 0; v < 8; v++) {
    int m = mBase + 8 * h + v;
    size_t off = (size_t)m * ldc + (nBase + n);
    float x = acc[v];
    if (mode == 1) x = 0.5f * (1.0f / (1.0f + expf(-x))) + aux[off];  // y_n = 0.5*sigmoid + feat
    else if (mode == 2) x = aux[off] - x;                             // z_n = z - A@z
    C[off] = x;
  }
}

// ---------------- generic bf16 WMMA GEMM: C[M,N] = A[M,K] * BT[N,K]^T, 32x32 tile/wave ----------------
__global__ __launch_bounds__(256) void gemm_wmma_bf16(
    const __bf16* __restrict__ A, int lda,
    const __bf16* __restrict__ BT, int ldbt,
    float* __restrict__ C, int Nn, int K,
    int mode, const float* __restrict__ aux) {
  int wave = (int)((blockIdx.x * blockDim.x + threadIdx.x) >> 5);
  int lane = threadIdx.x & 31;
  int tilesN = Nn >> 5;                 // 32-wide wave tiles across N
  int tm = wave / tilesN, tn = wave - tm * tilesN;
  int m0 = tm << 5, n0 = tn << 5;
  v8f a00 = {}, a01 = {}, a10 = {}, a11 = {};
  for (int kb = 0; kb < K; kb += 32) {
    v16bf af0 = load_a_frag(A, lda, m0,      kb, lane);
    v16bf af1 = load_a_frag(A, lda, m0 + 16, kb, lane);
    v16bf bf0 = load_b_frag(BT, ldbt, n0,      kb, lane);
    v16bf bf1 = load_b_frag(BT, ldbt, n0 + 16, kb, lane);
    a00 = __builtin_amdgcn_wmma_f32_16x16x32_bf16(false, af0, false, bf0, (short)0, a00, false, false);
    a01 = __builtin_amdgcn_wmma_f32_16x16x32_bf16(false, af0, false, bf1, (short)0, a01, false, false);
    a10 = __builtin_amdgcn_wmma_f32_16x16x32_bf16(false, af1, false, bf0, (short)0, a10, false, false);
    a11 = __builtin_amdgcn_wmma_f32_16x16x32_bf16(false, af1, false, bf1, (short)0, a11, false, false);
  }
  store_tile(C, Nn, m0,      n0,      a00, lane, mode, aux);
  store_tile(C, Nn, m0,      n0 + 16, a01, lane, mode, aux);
  store_tile(C, Nn, m0 + 16, n0,      a10, lane, mode, aux);
  store_tile(C, Nn, m0 + 16, n0 + 16, a11, lane, mode, aux);
}

// ---------------- one-time prep: masked_EA = 0.5*EA*w22; w1 -> bf16; EP0 = ones ----------------
__global__ __launch_bounds__(256) void prep_kernel(const float* __restrict__ EA,
                                                   const float* __restrict__ w22,
                                                   const float* __restrict__ w1,
                                                   float* __restrict__ mEA,
                                                   __bf16* __restrict__ w1b,
                                                   float* __restrict__ EP) {
  size_t idx = (size_t)blockIdx.x * blockDim.x + threadIdx.x;
  mEA[idx] = 0.5f * EA[idx] * w22[idx];
  w1b[idx] = f2bf(w1[idx]);
  EP[idx]  = 1.0f;
}

// ---------------- fp32 [4096,256] -> bf16 row-major (optional) + bf16 transpose [256,4096] ----------------
__global__ __launch_bounds__(256) void convT_kernel(const float* __restrict__ x,
                                                    __bf16* __restrict__ xrow,
                                                    __bf16* __restrict__ xT) {
  int idx = blockIdx.x * 256 + threadIdx.x;
  int i = idx >> 8, j = idx & 255;
  __bf16 b = f2bf(x[idx]);
  if (xrow) xrow[idx] = b;
  xT[(size_t)j * NNODE + i] = b;
}

// ---------------- fused elementwise: A = adj*EP + mEA (bf16); B = EP^T + A - yyT (bf16);
//                  EPnew = yyT - mEA (edge kernel then atomically subtracts 0.5*P) ----------------
__global__ __launch_bounds__(256) void build_ab_kernel(const float* __restrict__ EP,
                                                       const float* __restrict__ adj,
                                                       const float* __restrict__ mEA,
                                                       const float* __restrict__ yyT,
                                                       __bf16* __restrict__ Ab,
                                                       __bf16* __restrict__ Bb,
                                                       float* __restrict__ EPnew) {
  __shared__ float t[32][33];
  int tx = threadIdx.x, ty = threadIdx.y;
  int bi = blockIdx.y * 32, bj = blockIdx.x * 32;
  for (int r = ty; r < 32; r += 8)
    t[r][tx] = EP[(size_t)(bj + r) * NNODE + (bi + tx)];   // EP rows bj.., cols bi.. (coalesced)
  __syncthreads();
  for (int r = ty; r < 32; r += 8) {
    size_t idx = (size_t)(bi + r) * NNODE + (bj + tx);
    float ep = EP[idx];
    float me = mEA[idx];
    float a  = adj[idx] * ep + me;
    float yy = yyT[idx];
    Ab[idx] = f2bf(a);
    Bb[idx] = f2bf(t[tx][r] + a - yy);                     // t[tx][r] = EP[bj+tx][bi+r] = EP^T[i][j]
    EPnew[idx] = yy - me;
  }
}

// ---------------- edge scatter: one wave per edge; EPnew[i,j] -= 0.5 * ||z_i - z_j|| * val ----------------
__global__ __launch_bounds__(256) void edge_kernel(const float* __restrict__ z,
                                                   const int* __restrict__ ei,
                                                   const int* __restrict__ ej,
                                                   const float* __restrict__ ev,
                                                   float* __restrict__ EPnew) {
  int e = (int)((blockIdx.x * blockDim.x + threadIdx.x) >> 5);
  int lane = threadIdx.x & 31;
  int i = ei[e], j = ej[e];
  const float* zi = z + (size_t)i * NFEAT;
  const float* zj = z + (size_t)j * NFEAT;
  float s = 0.f;
#pragma unroll
  for (int k = lane; k < NFEAT; k += 32) { float d = zi[k] - zj[k]; s += d * d; }
  for (int off = 16; off > 0; off >>= 1) s += __shfl_down(s, off, 32);
  if (lane == 0)
    atomicAdd(&EPnew[(size_t)i * NNODE + j], -0.5f * sqrtf(s) * ev[e]);
}

// ---------------- head: row-normalize y,z; logits = [z|y] @ W + b; log_softmax ----------------
__global__ __launch_bounds__(256) void head_kernel(const float* __restrict__ Y,
                                                   const float* __restrict__ Z,
                                                   const float* __restrict__ W,
                                                   const float* __restrict__ bias,
                                                   float* __restrict__ out) {
  __shared__ float sh[512];
  __shared__ float lg[NCLS];
  int r = blockIdx.x, t = threadIdx.x;
  float zv = Z[(size_t)r * NFEAT + t];
  float yv = Y[(size_t)r * NFEAT + t];
  sh[t] = zv * zv; sh[256 + t] = yv * yv;
  __syncthreads();
  for (int s = 128; s > 0; s >>= 1) {
    if (t < s) { sh[t] += sh[t + s]; sh[256 + t] += sh[256 + t + s]; }
    __syncthreads();
  }
  float zn = fmaxf(sqrtf(sh[0]), 1e-12f);
  float yn = fmaxf(sqrtf(sh[256]), 1e-12f);
  __syncthreads();
  float pz = zv / zn, py = yv / yn;
  float acc[NCLS];
#pragma unroll
  for (int c = 0; c < NCLS; c++)
    acc[c] = pz * W[t * NCLS + c] + py * W[(256 + t) * NCLS + c];
  for (int c = 0; c < NCLS; c++) {
    sh[t] = acc[c];
    __syncthreads();
    for (int s = 128; s > 0; s >>= 1) {
      if (t < s) sh[t] += sh[t + s];
      __syncthreads();
    }
    if (t == 0) lg[c] = sh[0] + bias[c];
    __syncthreads();
  }
  if (t == 0) {
    float mx = lg[0];
    for (int c = 1; c < NCLS; c++) mx = fmaxf(mx, lg[c]);
    float se = 0.f;
    for (int c = 0; c < NCLS; c++) se += expf(lg[c] - mx);
    float lse = logf(se) + mx;
    for (int c = 0; c < NCLS; c++) out[(size_t)r * NCLS + c] = lg[c] - lse;
  }
}

extern "C" void kernel_launch(void* const* d_in, const int* in_sizes, int n_in,
                              void* d_out, int out_size, void* d_ws, size_t ws_size,
                              hipStream_t stream) {
  (void)in_sizes; (void)n_in; (void)out_size; (void)ws_size;
  const float* feat = (const float*)d_in[0];
  const float* z0   = (const float*)d_in[1];
  const float* w1   = (const float*)d_in[2];
  const float* EA   = (const float*)d_in[3];
  const float* w2w  = (const float*)d_in[4];
  const float* w2b  = (const float*)d_in[5];
  const float* adj  = (const float*)d_in[6];
  const float* w22  = (const float*)d_in[7];
  const float* ev   = (const float*)d_in[8];
  const int*   ei   = (const int*)d_in[9];
  const int*   ej   = (const int*)d_in[10];
  float* out = (float*)d_out;

  const size_t NN = (size_t)NNODE * NNODE;   // 16777216
  const size_t NF = (size_t)NNODE * NFEAT;   // 1048576

  char* p = (char*)d_ws;
  auto alloc = [&](size_t bytes) { char* r = p; p += (bytes + 255) & ~(size_t)255; return r; };
  float*  EPa  = (float*)alloc(NN * 4);      // 64MB
  float*  EPb  = (float*)alloc(NN * 4);      // 64MB
  float*  mEA  = (float*)alloc(NN * 4);      // 64MB
  float*  yyT  = (float*)alloc(NN * 4);      // 64MB
  __bf16* w1b  = (__bf16*)alloc(NN * 2);     // 32MB
  __bf16* Ab   = (__bf16*)alloc(NN * 2);     // 32MB
  __bf16* Bb   = (__bf16*)alloc(NN * 2);     // 32MB
  float*  Y0   = (float*)alloc(NF * 4);
  float*  Y1   = (float*)alloc(NF * 4);
  float*  Z0   = (float*)alloc(NF * 4);
  float*  Z1   = (float*)alloc(NF * 4);
  float*  TMP  = (float*)alloc(NF * 4);
  __bf16* yb   = (__bf16*)alloc(NF * 2);
  __bf16* ybT  = (__bf16*)alloc(NF * 2);
  __bf16* zbT  = (__bf16*)alloc(NF * 2);
  __bf16* tmpT = (__bf16*)alloc(NF * 2);

  prep_kernel<<<(int)(NN / 256), 256, 0, stream>>>(EA, w22, w1, mEA, w1b, EPa);
  hipMemcpyAsync(Y0, feat, NF * 4, hipMemcpyDeviceToDevice, stream);
  hipMemcpyAsync(Z0, z0,   NF * 4, hipMemcpyDeviceToDevice, stream);

  float* Ycur = Y0; float* Ynew = Y1;
  float* Zcur = Z0; float* Znew = Z1;
  float* EPcur = EPa; float* EPnew = EPb;

  const int convBlocks = (int)(NF / 256);
  const int wavesBig   = (NNODE / 32) * (NNODE / 32);   // yyT
  const int wavesTall  = (NNODE / 32) * (NFEAT / 32);   // [4096,4096]x[4096,256]

  for (int l = 0; l < NLAYER; l++) {
    convT_kernel<<<convBlocks, 256, 0, stream>>>(Ycur, yb, ybT);
    convT_kernel<<<convBlocks, 256, 0, stream>>>(Zcur, nullptr, zbT);

    // yyT = y @ y^T  (NT: BT operand is yb itself), K = 256
    gemm_wmma_bf16<<<wavesBig / 8, 256, 0, stream>>>(yb, NFEAT, yb, NFEAT,
                                                     yyT, NNODE, NFEAT, 0, nullptr);

    // A = adj*EP + mEA ; B = EP^T + A - yyT ; EPnew = yyT - mEA
    build_ab_kernel<<<dim3(NNODE / 32, NNODE / 32), dim3(32, 8), 0, stream>>>(
        EPcur, adj, mEA, yyT, Ab, Bb, EPnew);

    // EPnew[i,j] -= 0.5 * ||z_i - z_j|| * edge_val  (uses pre-update z)
    edge_kernel<<<NEDGE / 8, 256, 0, stream>>>(Zcur, ei, ej, ev, EPnew);

    // temp = B @ y, K = 4096
    gemm_wmma_bf16<<<wavesTall / 8, 256, 0, stream>>>(Bb, NNODE, ybT, NNODE,
                                                      TMP, NFEAT, NNODE, 0, nullptr);
    convT_kernel<<<convBlocks, 256, 0, stream>>>(TMP, nullptr, tmpT);

    // y_n = 0.5*sigmoid(w1 @ temp) + feat
    gemm_wmma_bf16<<<wavesTall / 8, 256, 0, stream>>>(w1b, NNODE, tmpT, NNODE,
                                                      Ynew, NFEAT, NNODE, 1, feat);
    // z_n = z - A @ z
    gemm_wmma_bf16<<<wavesTall / 8, 256, 0, stream>>>(Ab, NNODE, zbT, NNODE,
                                                      Znew, NFEAT, NNODE, 2, Zcur);

    float* t;
    t = Ycur; Ycur = Ynew; Ynew = t;
    t = Zcur; Zcur = Znew; Znew = t;
    t = EPcur; EPcur = EPnew; EPnew = t;
  }

  head_kernel<<<NNODE, 256, 0, stream>>>(Ycur, Zcur, w2w, w2b, out);
}
